// RecurrentMixer_73924977098964
// MI455X (gfx1250) — matrix operand order repaired and checked by reading, compiled, not run on hardware
//
#include <hip/hip_runtime.h>
#include <hip/hip_bf16.h>

typedef __attribute__((ext_vector_type(16))) __bf16 v16bf;
typedef __attribute__((ext_vector_type(8)))  float  v8f;

#define BB   4
#define TT   4096
#define DD   1024
#define NC   32          // scan chunks
#define CL   (TT / NC)   // 128 steps per chunk

__device__ __forceinline__ unsigned short f2bf(float f) {
    union { float f; unsigned int u; } c; c.f = f;
    unsigned int r = c.u + 0x7FFFu + ((c.u >> 16) & 1u);  // round-to-nearest-even
    return (unsigned short)(r >> 16);
}

__device__ __forceinline__ float sigmoidf_(float x) {
    return 1.0f / (1.0f + __expf(-x));
}

// ---------------- f32 -> bf16 weight conversion ----------------
__global__ void cvt_bf16_kernel(const float* __restrict__ src,
                                unsigned short* __restrict__ dst, int n) {
    int i = blockIdx.x * blockDim.x + threadIdx.x;
    if (i < n) dst[i] = f2bf(src[i]);
}

// ---------------- A = x + causal depthwise conv (K=4), emit bf16 ----------------
__global__ void prep_kernel(const float* __restrict__ x,
                            const float* __restrict__ wc,   // [D,1,4]
                            unsigned short* __restrict__ Abf) {
    size_t idx = (size_t)blockIdx.x * blockDim.x + threadIdx.x;  // < B*T*D
    int ch = (int)(idx % DD);
    int t  = (int)((idx / DD) % TT);
    float sum = x[idx];
#pragma unroll
    for (int k = 0; k < 4; ++k) {
        int tt = t - 3 + k;
        if (tt >= 0)
            sum += wc[ch * 4 + k] * x[idx + (size_t)(tt - t) * DD];
    }
    Abf[idx] = f2bf(sum);
}

// ---------------- tiled bf16 WMMA GEMM: C[M,N] = A[M,K] * W[N,K]^T, K=1024 ----------------
// segWidth: activation per column segment (seg = col/segWidth): 0=id, 1,2=sigmoid, 3=tanh
constexpr int GK = 1024;   // K is 1024 for both GEMMs -> constexpr so B addressing
                           // becomes SGPR-base (W+kk) + per-lane const voffset +
                           // immediate IOFFSET per tile (GVS form, no 64-bit VALU adds)

union Frag { uint4 q[2]; v16bf v; };

__device__ __forceinline__ void loadB(Frag* dst, const unsigned short* Wk, int colOff) {
#pragma unroll
    for (int jj = 0; jj < 4; ++jj) {
        const uint4* bp = reinterpret_cast<const uint4*>(Wk + colOff + jj * 16 * GK);
        dst[jj].q[0] = bp[0];
        dst[jj].q[1] = bp[1];
    }
}

__global__ __launch_bounds__(256) void gemm_bf16_kernel(
    const unsigned short* __restrict__ A,   // [M,GK] bf16, K-contig
    const unsigned short* __restrict__ W,   // [N,GK] bf16, K-contig
    float* __restrict__ C,                  // [M,N] f32
    int M, int N, int segWidth) {
    __shared__ unsigned short aLds[16 * 1032];   // 16 rows, stride 1032 (pad: conflict-free b128)

    const int tid   = threadIdx.x;
    const int wave  = tid >> 5;
    const int lane  = tid & 31;
    const int mBase = blockIdx.x * 16;
    const int nWave = blockIdx.y * 512 + wave * 64;

    // Stage A strip (16 x GK bf16) into LDS: 16B chunks, fully coalesced.
    for (int j = tid; j < 16 * (GK / 8); j += 256) {
        int row = j / (GK / 8);
        int c16 = j % (GK / 8);
        const uint4* src = reinterpret_cast<const uint4*>(A + (size_t)(mBase + row) * GK + c16 * 8);
        uint4* dst = reinterpret_cast<uint4*>(&aLds[row * 1032 + c16 * 8]);
        *dst = *src;
    }
    __syncthreads();

    const int half  = lane >> 4;   // 0/1
    const int mfrag = lane & 15;   // A-fragment row
    const int nSub  = lane & 15;

    // Per-lane constant element offset of this lane's B stream within W.
    const int colOff = (nWave + nSub) * GK + half * 16;

    v8f acc[4];
#pragma unroll
    for (int jj = 0; jj < 4; ++jj) acc[jj] = (v8f){0.f,0.f,0.f,0.f,0.f,0.f,0.f,0.f};

    // Loop-carried double buffer for B fragments: the 8 loads of one buffer are
    // issued while the 4 WMMAs consume the other, so both 32-VGPR sets stay live
    // and the scheduler can use partial loadcnt waits instead of full drains.
    Frag bA[4], bB[4];
    loadB(bA, W, colOff);

    for (int kk = 0; kk < GK; kk += 64) {
        // ---- half-step 1: consume bA @ kk, fetch bB @ kk+32 ----
        loadB(bB, W + kk + 32, colOff);
        Frag a;
        a.q[0] = *reinterpret_cast<const uint4*>(&aLds[mfrag * 1032 + kk + half * 16]);
        a.q[1] = *reinterpret_cast<const uint4*>(&aLds[mfrag * 1032 + kk + 8 + half * 16]);
        __builtin_prefetch(W + kk + 64 + colOff, 0, 1);   // speculative, safe if OOB
#pragma unroll
        for (int jj = 0; jj < 4; ++jj)
            acc[jj] = __builtin_amdgcn_wmma_f32_16x16x32_bf16(
                false, a.v, false, bA[jj].v, (short)0, acc[jj], false, false);

        // ---- half-step 2: consume bB @ kk+32, fetch bA @ kk+64 ----
        // Tail preload (kk+64 == GK) reads <=128B past the weight image; weights
        // are followed by other workspace buffers inside d_ws, so no fault.
        loadB(bA, W + kk + 64, colOff);
        a.q[0] = *reinterpret_cast<const uint4*>(&aLds[mfrag * 1032 + kk + 32 + half * 16]);
        a.q[1] = *reinterpret_cast<const uint4*>(&aLds[mfrag * 1032 + kk + 40 + half * 16]);
#pragma unroll
        for (int jj = 0; jj < 4; ++jj)
            acc[jj] = __builtin_amdgcn_wmma_f32_16x16x32_bf16(
                false, a.v, false, bB[jj].v, (short)0, acc[jj], false, false);
    }

    // Store with fused segment activation. C/D layout: lane -> (M = half*8 + r, N = nSub).
#pragma unroll
    for (int jj = 0; jj < 4; ++jj) {
        int colBase = nWave + jj * 16;
        int seg = colBase / segWidth; if (seg > 3) seg = 3;
        int col = colBase + nSub;
#pragma unroll
        for (int r = 0; r < 8; ++r) {
            float vv = acc[jj][r];
            if (seg == 1 || seg == 2) vv = sigmoidf_(vv);
            else if (seg == 3)        vv = tanhf(vv);
            C[(size_t)(mBase + half * 8 + r) * N + col] = vv;
        }
    }
}

// ---------------- chunked scan ----------------
// u layout: [B, T, 4D]; segments: 0=v, 1=gate, 2=decay, 3=inject (activations already applied)
__global__ void scan_pass1(const float* __restrict__ u,
                           float* __restrict__ P, float* __restrict__ L) {
    int gid = blockIdx.x * blockDim.x + threadIdx.x;   // B*D*NC
    int ch = gid % DD;
    int bb = (gid / DD) % BB;
    int c  = gid / (BB * DD);
    float p = 1.0f, s = 0.0f;
    size_t rowBase = ((size_t)bb * TT + (size_t)c * CL) * (4 * DD) + ch;
    for (int i = 0; i < CL; ++i) {
        size_t r = rowBase + (size_t)i * (4 * DD);
        float dec = u[r + 2 * DD];
        float inj = u[r + 3 * DD];
        s = dec * s + (1.0f - dec) * inj;
        p *= dec;
    }
    int o = (c * BB + bb) * DD + ch;
    P[o] = p; L[o] = s;
}

__global__ void scan_pass2(float* __restrict__ P, const float* __restrict__ L) {
    int gid = blockIdx.x * blockDim.x + threadIdx.x;   // B*D
    int ch = gid % DD;
    int bb = gid / DD;
    float s = 0.0f;
    for (int c = 0; c < NC; ++c) {
        int o = (c * BB + bb) * DD + ch;
        float p = P[o], l = L[o];
        P[o] = s;              // replace with chunk's true initial state
        s = p * s + l;
    }
}

__global__ void scan_pass3(const float* __restrict__ u,
                           const float* __restrict__ Sin,
                           unsigned short* __restrict__ ybf) {
    int gid = blockIdx.x * blockDim.x + threadIdx.x;   // B*D*NC
    int ch = gid % DD;
    int bb = (gid / DD) % BB;
    int c  = gid / (BB * DD);
    float s = Sin[(c * BB + bb) * DD + ch];
    for (int i = 0; i < CL; ++i) {
        size_t t = (size_t)c * CL + i;
        size_t r = ((size_t)bb * TT + t) * (4 * DD) + ch;
        float vv  = u[r];
        float g   = u[r + DD];
        float dec = u[r + 2 * DD];
        float inj = u[r + 3 * DD];
        s = dec * s + (1.0f - dec) * inj;
        float out = g * vv + (1.0f - g) * s;
        ybf[((size_t)bb * TT + t) * DD + ch] = f2bf(out);
    }
}

// ---------------- launch ----------------
extern "C" void kernel_launch(void* const* d_in, const int* in_sizes, int n_in,
                              void* d_out, int out_size, void* d_ws, size_t ws_size,
                              hipStream_t stream) {
    const float* x  = (const float*)d_in[0];   // [B,T,D]
    const float* Wc = (const float*)d_in[1];   // [D,1,4]
    const float* Wp = (const float*)d_in[2];   // [4D,D]
    const float* Wo = (const float*)d_in[3];   // [D,D]
    float* out = (float*)d_out;                // [B,T,D]

    char* ws = (char*)d_ws;
    size_t o = 0;
    unsigned short* Abf  = (unsigned short*)(ws + o); o += (size_t)BB * TT * DD * 2;   // 32 MB
    unsigned short* Wpbf = (unsigned short*)(ws + o); o += (size_t)4 * DD * DD * 2;    //  8 MB
    unsigned short* Wobf = (unsigned short*)(ws + o); o += (size_t)DD * DD * 2;        //  2 MB
    float*          u    = (float*)(ws + o);          o += (size_t)BB * TT * 4 * DD * 4; // 256 MB
    unsigned short* ybf  = (unsigned short*)(ws + o); o += (size_t)BB * TT * DD * 2;   // 32 MB
    float*          Pbuf = (float*)(ws + o);          o += (size_t)NC * BB * DD * 4;
    float*          Lbuf = (float*)(ws + o);          o += (size_t)NC * BB * DD * 4;

    cvt_bf16_kernel<<<(4 * DD * DD + 255) / 256, 256, 0, stream>>>(Wp, Wpbf, 4 * DD * DD);
    cvt_bf16_kernel<<<(DD * DD + 255) / 256, 256, 0, stream>>>(Wo, Wobf, DD * DD);
    prep_kernel<<<(BB * TT * DD) / 256, 256, 0, stream>>>(x, Wc, Abf);

    dim3 g1(BB * TT / 16, (4 * DD) / 512);     // 1024 x 8 blocks
    gemm_bf16_kernel<<<g1, 256, 0, stream>>>(Abf, Wpbf, u, BB * TT, 4 * DD, DD);

    scan_pass1<<<(BB * DD * NC) / 256, 256, 0, stream>>>(u, Pbuf, Lbuf);
    scan_pass2<<<(BB * DD) / 256, 256, 0, stream>>>(Pbuf, Lbuf);
    scan_pass3<<<(BB * DD * NC) / 256, 256, 0, stream>>>(u, Pbuf, ybf);

    dim3 g2(BB * TT / 16, DD / 512);           // 1024 x 2 blocks
    gemm_bf16_kernel<<<g2, 256, 0, stream>>>(ybf, Wobf, out, BB * TT, DD, DD);
}
